// MultiHeadAttention_76656576299333
// MI455X (gfx1250) — matrix operand order repaired and checked by reading, not compiled, not run on hardware
//
#include <hip/hip_runtime.h>
#include <hip/hip_bf16.h>

#define BB 2
#define TT 2048
#define CC 1024
#define HH 16
#define DD 64
#define MM (BB*TT)   // 4096 rows in all projections

typedef __attribute__((ext_vector_type(16))) _Float16 v16h;
typedef __attribute__((ext_vector_type(8)))  _Float16 v8h;
typedef __attribute__((ext_vector_type(8)))  float    v8f;
typedef __attribute__((ext_vector_type(4)))  float    v4f;
typedef __attribute__((ext_vector_type(4)))  unsigned int v4u;
typedef __attribute__((ext_vector_type(8)))  int      v8i;
typedef __attribute__((ext_vector_type(4)))  int      v4i;

__device__ __forceinline__ v8f wmma16(v16h a, v16h b, v8f c) {
  // D(f32 16x16) = A(f16 16x32) * B(f16 32x16) + C
  return __builtin_amdgcn_wmma_f32_16x16x32_f16(false, a, false, b, (short)0, c,
                                                false, false);
}

// ---------------------------------------------------------------------------
// TDM: async 2D tile load Global -> LDS (f16 elements). 6-arg builtin form.
// D# per CDNA5 ISA ch.8: group0 = {count, lds_addr, global_addr, type=2},
// group1 = {data_size=2B, tensor_dim0/1, tile_dim0/1, tensor_dim0_stride}.
// ---------------------------------------------------------------------------
__device__ __forceinline__ void tdm_load_2d_f16(unsigned lds_off, const void* gp,
                                                unsigned tensor_d0, unsigned tensor_d1,
                                                unsigned tile_d0, unsigned tile_d1,
                                                unsigned long long stride0)
{
  unsigned long long ga = (unsigned long long)(uintptr_t)gp;
  v4u g0;
  g0[0] = 1u;                                            // count=1 (valid user D#)
  g0[1] = lds_off;                                       // lds_addr (bytes)
  g0[2] = (unsigned)(ga & 0xFFFFFFFFu);                  // global_addr[31:0]
  g0[3] = (unsigned)((ga >> 32) & 0x1FFFFFFu) | (2u << 30); // addr[56:32] | type=2
  v8i g1;
  g1[0] = (int)(1u << 16);                               // data_size=1 (2 bytes)
  g1[1] = (int)((tensor_d0 & 0xFFFFu) << 16);            // tensor_dim0[15:0]
  g1[2] = (int)((tensor_d0 >> 16) | ((tensor_d1 & 0xFFFFu) << 16));
  g1[3] = (int)((tensor_d1 >> 16) | (tile_d0 << 16));    // tile_dim0
  g1[4] = (int)(tile_d1 & 0xFFFFu);                      // tile_dim1 (tile_dim2=0)
  g1[5] = (int)(stride0 & 0xFFFFFFFFull);                // tensor_dim0_stride[31:0]
  g1[6] = (int)((stride0 >> 32) & 0xFFFFull);            // stride[47:32]
  g1[7] = 0;
  v4i z4 = {0, 0, 0, 0};
  v8i z8 = {0, 0, 0, 0, 0, 0, 0, 0};
  __builtin_amdgcn_tensor_load_to_lds(g0, g1, z4, z4, z8, 0);
}

// ---------------------------------------------------------------------------
// One-shot fp32 -> f16 conversion (bandwidth-bound; removes ALL conversions
// from the GEMM hot loops). 8 elements per thread.
// ---------------------------------------------------------------------------
__global__ __launch_bounds__(256)
void cvt_f32_to_f16_kernel(const float* __restrict__ src,
                           _Float16* __restrict__ dst, int n8)
{
  const int i = blockIdx.x * 256 + threadIdx.x;
  if (i < n8) {
    v4f a = *(const v4f*)(src + (size_t)i * 8);
    v4f b = *(const v4f*)(src + (size_t)i * 8 + 4);
    v8h o;
    #pragma unroll
    for (int j = 0; j < 4; ++j) {
      o[j]     = (_Float16)a[j];
      o[4 + j] = (_Float16)b[j];
    }
    *(v8h*)(dst + (size_t)i * 8) = o;
  }
}

// ---------------------------------------------------------------------------
// QKV projection, pure f16 operands: Y = X @ W^T + bias.
// One wave owns a 16x64 tile; per k-step: 2x16B A loads + 4x32B B loads +
// 4 WMMAs, zero conversions.
//   dmajor==0 : Y[b][h][t][d]   (Q, K)
//   dmajor==1 : Y[b][h][d][t]   (V: makes PV B-operand loads contiguous)
// ---------------------------------------------------------------------------
__global__ __launch_bounds__(256)
void proj_qkv_kernel(const _Float16* __restrict__ X, const _Float16* __restrict__ W,
                     const float* __restrict__ bias, _Float16* __restrict__ Y,
                     int dmajor)
{
  const int lane = threadIdx.x & 31;
  const int wave = threadIdx.x >> 5;
  const int tile = blockIdx.x * 8 + wave;
  const int tilesN = CC / 64;                 // 16
  const int tileN = tile % tilesN;
  const int tileM = tile / tilesN;
  const int l15  = lane & 15;
  const int half = lane >> 4;
  const int kbA  = half * 16;
  const int kbB  = half * 16;

  const _Float16* Xr = X + (size_t)(tileM * 16 + l15) * CC;
  const _Float16* Wp[4];
  #pragma unroll
  for (int g = 0; g < 4; ++g)
    Wp[g] = W + (size_t)(tileN * 64 + g * 16 + l15) * CC;

  v8f acc0 = {}, acc1 = {}, acc2 = {}, acc3 = {};
  for (int k0 = 0; k0 < CC; k0 += 32) {
    // all loads of this k-step issued first (distinct regs -> load clause)
    v8h lo = *(const v8h*)(Xr + k0 + kbA);
    v8h hi = *(const v8h*)(Xr + k0 + 8 + kbA);
    v16h b0 = *(const v16h*)(Wp[0] + k0 + kbB);
    v16h b1 = *(const v16h*)(Wp[1] + k0 + kbB);
    v16h b2 = *(const v16h*)(Wp[2] + k0 + kbB);
    v16h b3 = *(const v16h*)(Wp[3] + k0 + kbB);
    if ((k0 & 127) == 0) {                     // one prefetch per 256B line
      __builtin_prefetch((const void*)(Xr + k0 + 512), 0, 3);
      #pragma unroll
      for (int g = 0; g < 4; ++g)
        __builtin_prefetch((const void*)(Wp[g] + k0 + 512), 0, 3);
    }
    v16h a;
    #pragma unroll
    for (int i = 0; i < 8; ++i) { a[i] = lo[i]; a[8 + i] = hi[i]; }
    acc0 = wmma16(a, b0, acc0);
    acc1 = wmma16(a, b1, acc1);
    acc2 = wmma16(a, b2, acc2);
    acc3 = wmma16(a, b3, acc3);
  }

  #pragma unroll
  for (int g = 0; g < 4; ++g) {
    const v8f acc = (g == 0) ? acc0 : (g == 1) ? acc1 : (g == 2) ? acc2 : acc3;
    const int n = tileN * 64 + g * 16 + l15;
    const float bval = bias[n];
    const int h = n >> 6;
    const int d = n & 63;
    #pragma unroll
    for (int i = 0; i < 8; ++i) {
      const int m  = tileM * 16 + i + half * 8;
      const int b_ = m >> 11;
      const int t  = m & (TT - 1);
      size_t idx = dmajor ? (((size_t)(b_ * HH + h) * DD + d) * TT + t)
                          : (((size_t)(b_ * HH + h) * TT + t) * DD + d);
      Y[idx] = (_Float16)(acc[i] + bval);
    }
  }
}

// ---------------------------------------------------------------------------
// Flash attention. Block = 4 waves = 4 consecutive 16-row q-tiles of one
// (b,h). K (32x64) and V (64x32) tiles per 32-key block staged into
// double-buffered LDS by the Tensor Data Mover (wave 0 issues, TENSORcnt
// waits), prefetching block i+1 while computing block i. Online softmax.
// ---------------------------------------------------------------------------
__global__ __launch_bounds__(128)
void attn_kernel(const _Float16* __restrict__ Qh, const _Float16* __restrict__ Kh,
                 const _Float16* __restrict__ Vt, _Float16* __restrict__ ctx)
{
  const int lane = threadIdx.x & 31;
  const int wave = threadIdx.x >> 5;
  const int bh    = blockIdx.x >> 5;                 // b*H + h  (32 blocks per bh)
  const int qtile = ((blockIdx.x & 31) << 2) + wave; // 0..127
  const int l15  = lane & 15;
  const int half = lane >> 4;
  const int kbA  = half * 16;
  const int kbB  = half * 16;

  __shared__ __align__(32) _Float16 ldsK[2][32 * 64];  // [key][d]
  __shared__ __align__(32) _Float16 ldsV[2][64 * 32];  // [d][key]
  __shared__ __align__(32) _Float16 pb[4][16 * 32];
  _Float16* pbuf = &pb[wave][0];

  const _Float16* Qp = Qh + (size_t)bh * TT * DD;
  const _Float16* Kp = Kh + (size_t)bh * TT * DD;
  const _Float16* Vp = Vt + (size_t)bh * DD * TT;      // [d][t]

  // Q A-fragments (d 0..31 / 32..63), pre-scaled by 1/sqrt(D)=0.125 (exact f16)
  v16h a0, a1;
  {
    const _Float16 qs = (_Float16)0.125f;
    const _Float16* qr = Qp + (size_t)(qtile * 16 + l15) * DD;
    v8h x0 = *(const v8h*)(qr + kbA);
    v8h x1 = *(const v8h*)(qr + 8 + kbA);
    v8h x2 = *(const v8h*)(qr + 32 + kbA);
    v8h x3 = *(const v8h*)(qr + 40 + kbA);
    #pragma unroll
    for (int i = 0; i < 8; ++i) {
      a0[i] = x0[i] * qs; a0[8 + i] = x1[i] * qs;
      a1[i] = x2[i] * qs; a1[8 + i] = x3[i] * qs;
    }
  }

  v8f o0 = {}, o1 = {}, o2 = {}, o3 = {};
  float runmax[8], lsum[8];
  #pragma unroll
  for (int i = 0; i < 8; ++i) { runmax[i] = -3.0e38f; lsum[i] = 0.f; }

  // TDM prologue: stage key-block 0 into buffer 0
  if (wave == 0) {
    tdm_load_2d_f16((unsigned)(uintptr_t)&ldsK[0][0], Kp, DD, TT, DD, 32, DD);
    tdm_load_2d_f16((unsigned)(uintptr_t)&ldsV[0][0], Vp, TT, DD, 32, DD, TT);
  }

  const int NIT = TT / 32;                      // 64 key blocks
  for (int it = 0; it < NIT; ++it) {
    const int kb  = it * 32;
    const int cur = it & 1;
    if (wave == 0) __builtin_amdgcn_s_wait_tensorcnt(0);
    __syncthreads();                            // tiles[cur] ready; tiles[cur^1] free
    if (wave == 0 && it + 1 < NIT) {            // prefetch next key block
      tdm_load_2d_f16((unsigned)(uintptr_t)&ldsK[cur ^ 1][0],
                      Kp + (size_t)(kb + 32) * DD, DD, TT, DD, 32, DD);
      tdm_load_2d_f16((unsigned)(uintptr_t)&ldsV[cur ^ 1][0],
                      Vp + (kb + 32), TT, DD, 32, DD, TT);
    }

    const _Float16* Kt = &ldsK[cur][0];
    const _Float16* Vb = &ldsV[cur][0];

    // ---- scores S(16x32) = Q . K^T (B operand from LDS) ----
    v8f s0 = {}, s1 = {};
    {
      v16h b;
      b = *(const v16h*)(Kt + (l15)      * DD + kbB);      s0 = wmma16(a0, b, s0);
      b = *(const v16h*)(Kt + (l15)      * DD + 32 + kbB); s0 = wmma16(a1, b, s0);
      b = *(const v16h*)(Kt + (16 + l15) * DD + kbB);      s1 = wmma16(a0, b, s1);
      b = *(const v16h*)(Kt + (16 + l15) * DD + 32 + kbB); s1 = wmma16(a1, b, s1);
    }

    // ---- online softmax over this 32-key block ----
    float rm[8], corr[8], rs[8];
    #pragma unroll
    for (int i = 0; i < 8; ++i) rm[i] = fmaxf(s0[i], s1[i]);
    #pragma unroll
    for (int off = 1; off < 16; off <<= 1) {
      #pragma unroll
      for (int i = 0; i < 8; ++i)
        rm[i] = fmaxf(rm[i], __shfl_xor(rm[i], off, 32));
    }
    #pragma unroll
    for (int i = 0; i < 8; ++i) {
      float nm = fmaxf(runmax[i], rm[i]);
      corr[i] = __expf(runmax[i] - nm);
      runmax[i] = nm;
      s0[i] = __expf(s0[i] - nm);
      s1[i] = __expf(s1[i] - nm);
      rs[i] = s0[i] + s1[i];
    }
    #pragma unroll
    for (int off = 1; off < 16; off <<= 1) {
      #pragma unroll
      for (int i = 0; i < 8; ++i)
        rs[i] += __shfl_xor(rs[i], off, 32);
    }
    #pragma unroll
    for (int i = 0; i < 8; ++i) {
      lsum[i] = lsum[i] * corr[i] + rs[i];
      o0[i] *= corr[i]; o1[i] *= corr[i];
      o2[i] *= corr[i]; o3[i] *= corr[i];
    }

    // ---- P: C/D layout -> A layout via per-wave LDS buffer ----
    #pragma unroll
    for (int i = 0; i < 8; ++i) {
      const int r = i + half * 8;
      pbuf[r * 32 + l15]      = (_Float16)s0[i];
      pbuf[r * 32 + 16 + l15] = (_Float16)s1[i];
    }
    v16h pa;
    {
      v8h plo = *(const v8h*)(pbuf + l15 * 32 + kbA);
      v8h phi = *(const v8h*)(pbuf + l15 * 32 + 8 + kbA);
      #pragma unroll
      for (int i = 0; i < 8; ++i) { pa[i] = plo[i]; pa[8 + i] = phi[i]; }
    }

    // ---- O += P(16x32) . V(32x64); V tile is d-major in LDS ----
    {
      v16h vb;
      vb = *(const v16h*)(Vb + (l15)      * 32 + kbB); o0 = wmma16(pa, vb, o0);
      vb = *(const v16h*)(Vb + (16 + l15) * 32 + kbB); o1 = wmma16(pa, vb, o1);
      vb = *(const v16h*)(Vb + (32 + l15) * 32 + kbB); o2 = wmma16(pa, vb, o2);
      vb = *(const v16h*)(Vb + (48 + l15) * 32 + kbB); o3 = wmma16(pa, vb, o3);
    }
  }

  // ---- epilogue: normalize, store context [b][t][h*64+d] as f16 ----
  const int b_ = bh >> 4;
  const int h  = bh & 15;
  #pragma unroll
  for (int i = 0; i < 8; ++i) {
    const float inv = 1.0f / lsum[i];
    const int t = qtile * 16 + i + half * 8;
    _Float16* dst = ctx + (size_t)(b_ * TT + t) * CC + h * DD + l15;
    dst[0]  = (_Float16)(o0[i] * inv);
    dst[16] = (_Float16)(o1[i] * inv);
    dst[32] = (_Float16)(o2[i] * inv);
    dst[48] = (_Float16)(o3[i] * inv);
  }
}

// ---------------------------------------------------------------------------
// Output projection, pure f16 operands: out = ctx @ Wo^T + bo, fp32 output.
// ---------------------------------------------------------------------------
__global__ __launch_bounds__(256)
void out_proj_kernel(const _Float16* __restrict__ Xh, const _Float16* __restrict__ W,
                     const float* __restrict__ bias, float* __restrict__ Y)
{
  const int lane = threadIdx.x & 31;
  const int wave = threadIdx.x >> 5;
  const int tile = blockIdx.x * 8 + wave;
  const int tilesN = CC / 64;
  const int tileN = tile % tilesN;
  const int tileM = tile / tilesN;
  const int l15  = lane & 15;
  const int half = lane >> 4;
  const int kbA  = half * 16;
  const int kbB  = half * 16;

  const _Float16* Xr = Xh + (size_t)(tileM * 16 + l15) * CC;
  const _Float16* Wp[4];
  #pragma unroll
  for (int g = 0; g < 4; ++g)
    Wp[g] = W + (size_t)(tileN * 64 + g * 16 + l15) * CC;

  v8f acc0 = {}, acc1 = {}, acc2 = {}, acc3 = {};
  for (int k0 = 0; k0 < CC; k0 += 32) {
    v8h lo = *(const v8h*)(Xr + k0 + kbA);
    v8h hi = *(const v8h*)(Xr + k0 + 8 + kbA);
    v16h b0 = *(const v16h*)(Wp[0] + k0 + kbB);
    v16h b1 = *(const v16h*)(Wp[1] + k0 + kbB);
    v16h b2 = *(const v16h*)(Wp[2] + k0 + kbB);
    v16h b3 = *(const v16h*)(Wp[3] + k0 + kbB);
    if ((k0 & 127) == 0) {
      __builtin_prefetch((const void*)(Xr + k0 + 512), 0, 3);
      #pragma unroll
      for (int g = 0; g < 4; ++g)
        __builtin_prefetch((const void*)(Wp[g] + k0 + 512), 0, 3);
    }
    v16h a;
    #pragma unroll
    for (int i = 0; i < 8; ++i) { a[i] = lo[i]; a[8 + i] = hi[i]; }
    acc0 = wmma16(a, b0, acc0);
    acc1 = wmma16(a, b1, acc1);
    acc2 = wmma16(a, b2, acc2);
    acc3 = wmma16(a, b3, acc3);
  }

  #pragma unroll
  for (int g = 0; g < 4; ++g) {
    const v8f acc = (g == 0) ? acc0 : (g == 1) ? acc1 : (g == 2) ? acc2 : acc3;
    const int n = tileN * 64 + g * 16 + l15;
    const float bval = bias[n];
    #pragma unroll
    for (int i = 0; i < 8; ++i) {
      const int m = tileM * 16 + i + half * 8;
      Y[(size_t)m * CC + n] = acc[i] + bval;
    }
  }
}

// ---------------------------------------------------------------------------
extern "C" void kernel_launch(void* const* d_in, const int* in_sizes, int n_in,
                              void* d_out, int out_size, void* d_ws, size_t ws_size,
                              hipStream_t stream)
{
  (void)in_sizes; (void)n_in; (void)out_size; (void)ws_size;
  const float* q  = (const float*)d_in[0];
  const float* k  = (const float*)d_in[1];
  const float* v  = (const float*)d_in[2];
  const float* Wq = (const float*)d_in[3];
  const float* bq = (const float*)d_in[4];
  const float* Wk = (const float*)d_in[5];
  const float* bk = (const float*)d_in[6];
  const float* Wv = (const float*)d_in[7];
  const float* bv = (const float*)d_in[8];
  const float* Wo = (const float*)d_in[9];
  const float* bo = (const float*)d_in[10];
  float* out = (float*)d_out;

  const size_t elems  = (size_t)BB * TT * CC;     // 4M elements
  const size_t welems = (size_t)CC * CC;          // 1M elements
  _Float16* Qh  = (_Float16*)d_ws;                // head-split Q      (8 MB)
  _Float16* Kh  = Qh  + elems;                    // head-split K      (8 MB)
  _Float16* Vt  = Kh  + elems;                    // head-split V, d-major (8 MB)
  _Float16* ctx = Vt  + elems;                    // attention output  (8 MB)
  _Float16* qf  = ctx + elems;                    // f16 copies of inputs
  _Float16* kf  = qf  + elems;
  _Float16* vf  = kf  + elems;
  _Float16* Wqf = vf  + elems;                    // f16 copies of weights
  _Float16* Wkf = Wqf + welems;
  _Float16* Wvf = Wkf + welems;
  _Float16* Wof = Wvf + welems;

  // ---- one-shot f32 -> f16 conversions (bandwidth-bound) ----
  {
    const int n8i = (int)(elems / 8);             // 512K
    const int n8w = (int)(welems / 8);            // 128K
    dim3 tb(256);
    hipLaunchKernelGGL(cvt_f32_to_f16_kernel, dim3((n8i + 255) / 256), tb, 0, stream, q,  qf,  n8i);
    hipLaunchKernelGGL(cvt_f32_to_f16_kernel, dim3((n8i + 255) / 256), tb, 0, stream, k,  kf,  n8i);
    hipLaunchKernelGGL(cvt_f32_to_f16_kernel, dim3((n8i + 255) / 256), tb, 0, stream, v,  vf,  n8i);
    hipLaunchKernelGGL(cvt_f32_to_f16_kernel, dim3((n8w + 255) / 256), tb, 0, stream, Wq, Wqf, n8w);
    hipLaunchKernelGGL(cvt_f32_to_f16_kernel, dim3((n8w + 255) / 256), tb, 0, stream, Wk, Wkf, n8w);
    hipLaunchKernelGGL(cvt_f32_to_f16_kernel, dim3((n8w + 255) / 256), tb, 0, stream, Wv, Wvf, n8w);
    hipLaunchKernelGGL(cvt_f32_to_f16_kernel, dim3((n8w + 255) / 256), tb, 0, stream, Wo, Wof, n8w);
  }

  const int tiles = (MM / 16) * (CC / 64);        // 4096 wave-tiles
  dim3 gblk(tiles / 8), tblk(256);
  hipLaunchKernelGGL(proj_qkv_kernel, gblk, tblk, 0, stream, qf, Wqf, bq, Qh, 0);
  hipLaunchKernelGGL(proj_qkv_kernel, gblk, tblk, 0, stream, kf, Wkf, bk, Kh, 0);
  hipLaunchKernelGGL(proj_qkv_kernel, gblk, tblk, 0, stream, vf, Wvf, bv, Vt, 1);

  dim3 ga(BB * HH * (TT / 16) / 4);               // 1024 blocks, 4 waves each
  hipLaunchKernelGGL(attn_kernel, ga, dim3(128), 0, stream, Qh, Kh, Vt, ctx);

  hipLaunchKernelGGL(out_proj_kernel, gblk, tblk, 0, stream, ctx, Wof, bo, out);
}
